// TLSTMLayer_30657476559122
// MI455X (gfx1250) — compile-verified, matrix operand order checked by reading, not run on hardware
//
#include <hip/hip_runtime.h>
#include <hip/hip_bf16.h>
#include <math.h>

typedef __attribute__((ext_vector_type(16))) _Float16 v16h;
typedef __attribute__((ext_vector_type(8)))  float    v8f;

constexpr int SEQ = 4096;   // S
constexpr int HD  = 512;    // H
constexpr int HHD = 256;    // HH
constexpr int NG  = 2048;   // 2 dirs * 4 gates * HH

// CDNA5 16-bit A/B fragment K mapping (ISA 7.12.2):
//   element e of lane L holds K = (e&7) + (e>>3)*16 + (L>>4)*8
// => elements 0..7  = K  base..base+7   (contiguous)
//    elements 8..15 = K  base+16..+23   (contiguous), base = (L>>4)*8

__device__ __forceinline__ void load_a_frag(const float* __restrict__ base,
                                            int koff, int lane, v16h& a) {
    const int b0 = koff + ((lane >> 4) << 3);
    const float4 p0 = *reinterpret_cast<const float4*>(base + b0);
    const float4 p1 = *reinterpret_cast<const float4*>(base + b0 + 4);
    const float4 p2 = *reinterpret_cast<const float4*>(base + b0 + 16);
    const float4 p3 = *reinterpret_cast<const float4*>(base + b0 + 20);
    a[0]=(_Float16)p0.x;  a[1]=(_Float16)p0.y;  a[2]=(_Float16)p0.z;  a[3]=(_Float16)p0.w;
    a[4]=(_Float16)p1.x;  a[5]=(_Float16)p1.y;  a[6]=(_Float16)p1.z;  a[7]=(_Float16)p1.w;
    a[8]=(_Float16)p2.x;  a[9]=(_Float16)p2.y;  a[10]=(_Float16)p2.z; a[11]=(_Float16)p2.w;
    a[12]=(_Float16)p3.x; a[13]=(_Float16)p3.y; a[14]=(_Float16)p3.z; a[15]=(_Float16)p3.w;
}

// bcol points at column `col` of a row-major (K x ldb) matrix.
__device__ __forceinline__ void load_b_frag(const float* __restrict__ bcol,
                                            int koff, int lane, int ldb, v16h& b) {
    const int kb = koff + ((lane >> 4) << 3);
    #pragma unroll
    for (int e = 0; e < 8; ++e) b[e]     = (_Float16)bcol[(size_t)(kb + e) * ldb];
    #pragma unroll
    for (int e = 0; e < 8; ++e) b[e + 8] = (_Float16)bcol[(size_t)(kb + 16 + e) * ldb];
}

// ---------------------------------------------------------------------------
// Kernel 1: x_pre = concat(inputs[0], note_emb[nid], chunk_emb[cid]) @ W_comb + b_comb
// M=SEQ, K=1536 (3 x 512 segments), N=HD.
// Block = 128 threads (4 waves); each wave computes a 64M x 16N stripe:
// one B fragment feeds 4 WMMAs (B-column loads amortized 4x).
// ---------------------------------------------------------------------------
__global__ __launch_bounds__(128)
void k_front_gemm(const float* __restrict__ inputs,     // (B,S,H), batch 0 used
                  const int*   __restrict__ nids,       // (B,S), batch 0
                  const int*   __restrict__ cids,       // (B,S), batch 0
                  const float* __restrict__ note_emb,   // (512,H)
                  const float* __restrict__ chunk_emb,  // (512,H)
                  const float* __restrict__ W_comb,     // (1536,HD)
                  const float* __restrict__ b_comb,     // (HD)
                  float*       __restrict__ xpre)       // (SEQ,HD)
{
    const int lane = threadIdx.x & 31;
    const int wave = threadIdx.x >> 5;
    const int m0   = blockIdx.x * 64;
    const int n0   = blockIdx.y * 64 + wave * 16;
    const int col  = n0 + (lane & 15);

    int rows[4], nid[4], cid[4];
    #pragma unroll
    for (int mi = 0; mi < 4; ++mi) {
        rows[mi] = m0 + mi * 16 + (lane & 15);
        nid[mi]  = nids[rows[mi]];
        cid[mi]  = cids[rows[mi]];
    }

    v8f acc[4] = {};
    for (int seg = 0; seg < 3; ++seg) {               // concat segments, each 512 wide
        const float* abase[4];
        #pragma unroll
        for (int mi = 0; mi < 4; ++mi) {
            abase[mi] = (seg == 0) ? inputs    + (size_t)rows[mi] * HD
                      : (seg == 1) ? note_emb  + (size_t)nid[mi]  * HD
                                   : chunk_emb + (size_t)cid[mi]  * HD;
        }
        const float* bseg = W_comb + (size_t)seg * HD * HD + col;
        for (int kt = 0; kt < HD; kt += 32) {
            v16h bf;
            load_b_frag(bseg, kt, lane, HD, bf);
            #pragma unroll
            for (int mi = 0; mi < 4; ++mi) {
                v16h af;
                load_a_frag(abase[mi], kt, lane, af);
                acc[mi] = __builtin_amdgcn_wmma_f32_16x16x32_f16(
                    false, af, false, bf, (short)0, acc[mi], false, false);
            }
        }
    }
    const float bias = b_comb[col];
    #pragma unroll
    for (int mi = 0; mi < 4; ++mi)
        #pragma unroll
        for (int r = 0; r < 8; ++r) {
            const int m = m0 + mi * 16 + r + ((lane >> 4) << 3);
            xpre[(size_t)m * HD + col] = acc[mi][r] + bias;
        }
}

// ---------------------------------------------------------------------------
// Kernel 2: in-place LayerNorm over HD per row, then *ln_g + ln_b.
// ---------------------------------------------------------------------------
__global__ __launch_bounds__(256)
void k_layernorm(float* __restrict__ x,
                 const float* __restrict__ ln_g,
                 const float* __restrict__ ln_b)
{
    __shared__ float red[256];
    const int s = blockIdx.x;
    const int j = threadIdx.x;
    float v0 = x[(size_t)s * HD + j];
    float v1 = x[(size_t)s * HD + 256 + j];
    red[j] = v0 + v1;
    __syncthreads();
    for (int off = 128; off > 0; off >>= 1) {
        if (j < off) red[j] += red[j + off];
        __syncthreads();
    }
    const float mu = red[0] * (1.0f / HD);
    __syncthreads();
    const float d0 = v0 - mu, d1 = v1 - mu;
    red[j] = d0 * d0 + d1 * d1;
    __syncthreads();
    for (int off = 128; off > 0; off >>= 1) {
        if (j < off) red[j] += red[j + off];
        __syncthreads();
    }
    const float rstd = rsqrtf(red[0] * (1.0f / HD) + 1e-12f);
    x[(size_t)s * HD + j]       = d0 * rstd * ln_g[j]       + ln_b[j];
    x[(size_t)s * HD + 256 + j] = d1 * rstd * ln_g[256 + j] + ln_b[256 + j];
}

// ---------------------------------------------------------------------------
// Kernel 3: gx[s][d*1024+g*256+j] = x[s] @ Wx[d][g] + bg[d][g]
// M=SEQ, K=HD, N=NG.  Same 64M x 16N per-wave stripe.
// ---------------------------------------------------------------------------
__global__ __launch_bounds__(128)
void k_gate_gemm(const float* __restrict__ x,    // (SEQ,HD)
                 const float* __restrict__ Wx,   // (2,4,HD,HHD)
                 const float* __restrict__ bg,   // (2,4,HHD)
                 float*       __restrict__ gx)   // (SEQ,NG)
{
    const int lane = threadIdx.x & 31;
    const int wave = threadIdx.x >> 5;
    const int m0   = blockIdx.x * 64;
    const int n0   = blockIdx.y * 64 + wave * 16;
    const int ncol = n0 + (lane & 15);
    const int dg   = ncol >> 8;        // d*4+g (256-aligned blocks; a 16-tile never straddles)
    const int jcol = ncol & 255;

    const float* abase[4];
    #pragma unroll
    for (int mi = 0; mi < 4; ++mi)
        abase[mi] = x + (size_t)(m0 + mi * 16 + (lane & 15)) * HD;
    const float* bseg = Wx + (size_t)dg * HD * HHD + jcol;

    v8f acc[4] = {};
    for (int kt = 0; kt < HD; kt += 32) {
        v16h bf;
        load_b_frag(bseg, kt, lane, HHD, bf);
        #pragma unroll
        for (int mi = 0; mi < 4; ++mi) {
            v16h af;
            load_a_frag(abase[mi], kt, lane, af);
            acc[mi] = __builtin_amdgcn_wmma_f32_16x16x32_f16(
                false, af, false, bf, (short)0, acc[mi], false, false);
        }
    }
    const float bias = bg[dg * HHD + jcol];
    #pragma unroll
    for (int mi = 0; mi < 4; ++mi)
        #pragma unroll
        for (int r = 0; r < 8; ++r) {
            const int m = m0 + mi * 16 + r + ((lane >> 4) << 3);
            gx[(size_t)m * NG + ncol] = acc[mi][r] + bias;
        }
}

// ---------------------------------------------------------------------------
// Kernel 4: serial TLSTM scan, one block per direction, 1024 threads.
// Thread (j, kq): j = output unit 0..255, kq = K-slice 0..3 (64 k's each).
// h/c replicated across the 4 kq slices of the same j.
// ---------------------------------------------------------------------------
__global__ __launch_bounds__(1024)
void k_scan(const float* __restrict__ times,  // (B,S,1), batch 0 -> times[s]
            const float* __restrict__ gx,     // (SEQ,NG)
            const float* __restrict__ Wh,     // (2,4,HHD,HHD)
            const float* __restrict__ Wd,     // (2,HHD,HHD)
            const float* __restrict__ bd,     // (2,HHD)
            float*       __restrict__ finals) // (512)
{
    const int dir = blockIdx.x;
    const int tid = threadIdx.x;
    const int j   = tid & 255;
    const int kq  = tid >> 8;                  // 0..3
    const int kbeg = kq * 64, kend = kbeg + 64;

    const float* WhD = Wh + (size_t)dir * 4 * HHD * HHD;
    const float* WdD = Wd + (size_t)dir * HHD * HHD;
    const float  bdj = bd[dir * HHD + j];

    __shared__ float sh[HHD];
    __shared__ float sc[HHD];
    __shared__ float pd[4][HHD];               // Wd partials [kq][j]
    __shared__ float pg[4][4][HHD];            // gate partials [g][kq][j]

    float h = 0.0f, c = 0.0f;
    const float E = 2.718281828459045f;

    for (int step = 0; step < SEQ; ++step) {
        const int s = (dir == 0) ? step : (SEQ - 1 - step);
        const float t = (dir == 0) ? times[step]
                                   : (step == 0 ? times[0] : times[SEQ - step]);
        const float T = 1.0f / logf(t + E);

        if (kq == 0) { sh[j] = h; sc[j] = c; }
        __syncthreads();

        float ad = 0.0f, a0 = 0.0f, a1 = 0.0f, a2 = 0.0f, a3 = 0.0f;
        #pragma unroll 4
        for (int k = kbeg; k < kend; ++k) {
            const float hk = sh[k];
            const float ck = sc[k];
            ad += ck * WdD[k * HHD + j];
            a0 += hk * WhD[(0 * HHD + k) * HHD + j];
            a1 += hk * WhD[(1 * HHD + k) * HHD + j];
            a2 += hk * WhD[(2 * HHD + k) * HHD + j];
            a3 += hk * WhD[(3 * HHD + k) * HHD + j];
        }
        pd[kq][j]    = ad;
        pg[0][kq][j] = a0;
        pg[1][kq][j] = a1;
        pg[2][kq][j] = a2;
        pg[3][kq][j] = a3;
        if (step + 1 < SEQ) {
            const int sn = (dir == 0) ? step + 1 : (SEQ - 2 - step);
            __builtin_prefetch(&gx[(size_t)sn * NG + dir * 1024 + j], 0, 1);
        }
        __syncthreads();

        const size_t gbase = (size_t)s * NG + dir * 1024 + j;
        const float sd = bdj + pd[0][j] + pd[1][j] + pd[2][j] + pd[3][j];
        const float g0 = gx[gbase]       + pg[0][0][j] + pg[0][1][j] + pg[0][2][j] + pg[0][3][j];
        const float g1 = gx[gbase + 256] + pg[1][0][j] + pg[1][1][j] + pg[1][2][j] + pg[1][3][j];
        const float g2 = gx[gbase + 512] + pg[2][0][j] + pg[2][1][j] + pg[2][2][j] + pg[2][3][j];
        const float g3 = gx[gbase + 768] + pg[3][0][j] + pg[3][1][j] + pg[3][2][j] + pg[3][3][j];

        const float cst   = tanhf(sd);
        const float c_adj = c - cst + T * cst;
        const float gi = 1.0f / (1.0f + __expf(-g0));   // i
        const float gf = 1.0f / (1.0f + __expf(-g1));   // f
        const float go = 1.0f / (1.0f + __expf(-g2));   // o
        const float gc = 1.0f / (1.0f + __expf(-g3));   // cg
        const float ct = gf * c_adj + gi * gc;
        h = go * tanhf(ct);
        c = ct;
        // pd/pg (read above) are rewritten only after the next barrier; sh/sc
        // writes race with nothing (all readers passed the barrier).
    }
    if (kq == 0) finals[dir * HHD + j] = h;
}

// ---------------------------------------------------------------------------
// Kernel 5: logits = concat(h_f, h_b) @ W_cls + b_cls; guarded sigmoid.
// ---------------------------------------------------------------------------
__global__ __launch_bounds__(256)
void k_head(const float* __restrict__ finals,  // (512)
            const float* __restrict__ W_cls,   // (512,1)
            const float* __restrict__ b_cls,   // (1)
            float*       __restrict__ out)
{
    __shared__ float red[256];
    const int j = threadIdx.x;
    red[j] = finals[j] * W_cls[j] + finals[256 + j] * W_cls[256 + j];
    __syncthreads();
    for (int off = 128; off > 0; off >>= 1) {
        if (j < off) red[j] += red[j + off];
        __syncthreads();
    }
    if (j == 0) {
        float l = red[0] + b_cls[0];
        if (isnan(l) || isinf(l)) l = 0.0f;
        float p = 1.0f / (1.0f + expf(-l));
        if (isnan(p) || isinf(p)) p = 0.0f;
        out[0] = p;
    }
}

extern "C" void kernel_launch(void* const* d_in, const int* in_sizes, int n_in,
                              void* d_out, int out_size, void* d_ws, size_t ws_size,
                              hipStream_t stream) {
    const float* inputs    = (const float*)d_in[0];
    const float* times     = (const float*)d_in[1];
    const int*   nids      = (const int*)  d_in[2];
    const int*   cids      = (const int*)  d_in[3];
    const float* note_emb  = (const float*)d_in[4];
    const float* chunk_emb = (const float*)d_in[5];
    const float* W_comb    = (const float*)d_in[6];
    const float* b_comb    = (const float*)d_in[7];
    const float* ln_g      = (const float*)d_in[8];
    const float* ln_b      = (const float*)d_in[9];
    const float* Wx        = (const float*)d_in[10];
    const float* Wh        = (const float*)d_in[11];
    const float* bg        = (const float*)d_in[12];
    const float* Wd        = (const float*)d_in[13];
    const float* bd        = (const float*)d_in[14];
    const float* W_cls     = (const float*)d_in[15];
    const float* b_cls     = (const float*)d_in[16];

    float* xbuf   = (float*)d_ws;                       // SEQ*HD
    float* gxbuf  = xbuf  + (size_t)SEQ * HD;           // SEQ*NG
    float* finals = gxbuf + (size_t)SEQ * NG;           // 512

    k_front_gemm<<<dim3(SEQ / 64, HD / 64), 128, 0, stream>>>(
        inputs, nids, cids, note_emb, chunk_emb, W_comb, b_comb, xbuf);
    k_layernorm<<<SEQ, 256, 0, stream>>>(xbuf, ln_g, ln_b);
    k_gate_gemm<<<dim3(SEQ / 64, NG / 64), 128, 0, stream>>>(
        xbuf, Wx, bg, gxbuf);
    k_scan<<<2, 1024, 0, stream>>>(times, gxbuf, Wh, Wd, bd, finals);
    k_head<<<1, 256, 0, stream>>>(finals, W_cls, b_cls, (float*)d_out);
}
